// RNNQNetwork_64072322122463
// MI455X (gfx1250) — matrix operand order repaired and checked by reading, not compiled, not run on hardware
//
#include <hip/hip_runtime.h>
#include <hip/hip_bf16.h>

#define T_STEPS 128
#define BATCH   512
#define HID     512
#define OBSD    512
#define ACTD    18
#define IN0     (OBSD + ACTD)
#define LN_EPS  1e-6f
#define KPAD    40    // padded K stride (halves) for 32-K LDS weight tiles
#define WFK     520   // padded K stride (halves) for resident full-K weight slice
#define W0K     544   // padded row length (halves) of transposed w0 (530 -> 544)
#define GRU_NWG 128   // workgroups in persistent GRU scan (16 Btiles x 8 Htiles)

typedef __bf16 bf16;
typedef __attribute__((ext_vector_type(16))) __bf16 v16bf;
typedef __attribute__((ext_vector_type(8)))  __bf16 v8bf;
typedef __attribute__((ext_vector_type(8)))  float  v8f;

// ---------------------------------------------------------------------------
// WMMA helpers (wave32, D[16x16,f32] = A[16x32,bf16] x B[32x16,bf16] + C)
// ---------------------------------------------------------------------------
static __device__ __forceinline__ v8f vzero8() {
  v8f v;
#pragma unroll
  for (int i = 0; i < 8; ++i) v[i] = 0.0f;
  return v;
}

static __device__ __forceinline__ v16bf cat16(v8bf lo, v8bf hi) {
  return __builtin_shufflevector(lo, hi, 0, 1, 2, 3, 4, 5, 6, 7,
                                 8, 9, 10, 11, 12, 13, 14, 15);
}

// A source row-major [row][k], 16B-aligned rows; stride in halves (mult of 8)
static __device__ __forceinline__ v16bf load_a_frag(const bf16* base, int lane, int stride) {
  const int r = lane & 15;
  const int koff = (lane >> 4) << 3;
  const bf16* p = base + (size_t)r * stride + koff;
  v8bf lo = *(const v8bf*)(p);
  v8bf hi = *(const v8bf*)(p + 16);
  return cat16(lo, hi);
}

// B source transposed [col][k], stride in halves (mult of 8)
static __device__ __forceinline__ v16bf load_b_frag_tr(const bf16* baseT, int lane, int stride) {
  const int c = lane & 15;
  const int kb = (lane >> 4) << 3;
  const bf16* p = baseT + (size_t)c * stride + kb;
  v8bf lo = *(const v8bf*)(p);
  v8bf hi = *(const v8bf*)(p + 16);
  return cat16(lo, hi);
}

static __device__ __forceinline__ v8f wmma_bf16(v16bf a, v16bf b, v8f c) {
  return __builtin_amdgcn_wmma_f32_16x16x32_bf16(false, a, false, b, (short)0, c, false, false);
}

static __device__ __forceinline__ float sigmoidf_(float x) {
  return 1.0f / (1.0f + __expf(-x));
}

static __device__ __forceinline__ v8bf pack8(const float* f) {
  v8bf p;
#pragma unroll
  for (int e = 0; e < 8; ++e) p[e] = (bf16)f[e];
  return p;
}

// ---------------------------------------------------------------------------
// Utility kernels
// ---------------------------------------------------------------------------
__global__ void copy_f32_kernel(const float* __restrict__ src, float* __restrict__ dst, int n) {
  int i = blockIdx.x * blockDim.x + threadIdx.x;
  if (i < n) dst[i] = src[i];
}

__global__ void zero_u32_kernel(unsigned* p) {
  if (threadIdx.x == 0 && blockIdx.x == 0) *p = 0u;
}

// dst[n*krow + k] = bf16(src[k*512 + n])  (transpose; N fixed at 512)
__global__ void cvt_tr_kernel(const float* __restrict__ src, bf16* __restrict__ dst,
                              int K, int krow) {
  int i = blockIdx.x * blockDim.x + threadIdx.x;
  if (i < K * 512) {
    int k = i >> 9, n = i & 511;
    dst[(size_t)n * krow + k] = (bf16)src[(size_t)k * 512 + n];
  }
}

// ---------------------------------------------------------------------------
// Encoder: per 32-row block of flattened [T*B]:
//   h0 = relu(LN(concat(obs,onehot) @ w0 + b0)); h1 = relu(LN(h0 @ w1 + b1));
//   gi_g = h1 @ w_ig + b_ig -> bf16 workspace.
// ---------------------------------------------------------------------------
#define ENC_A0_OFF   0                              // 32x32 bf16      = 2048
#define ENC_WC_OFF   2048                           // 512xKPAD bf16   = 40960
#define ENC_F32_OFF  (ENC_WC_OFF + 512 * KPAD * 2)  // 32x512 f32      = 65536
#define ENC_HB_OFF   (ENC_F32_OFF + 65536)          // 32x512 bf16     = 32768
#define ENC_PART_OFF (ENC_HB_OFF + 32768)           // 32x8x2 f32      = 2048
#define ENC_STAT_OFF (ENC_PART_OFF + 2048)          // 32x2 f32        = 256
#define ENC_SMEM_BYTES (ENC_STAT_OFF + 256)

static __device__ void ln_relu_block(float* FB, bf16* HB, float* PART, float* STAT,
                                     const float* __restrict__ s, const float* __restrict__ b,
                                     int tid) {
  const int r = tid >> 3, p = tid & 7;
  float sum = 0.0f, sq = 0.0f;
  const float* row = FB + r * 512;
  for (int c = p * 64; c < p * 64 + 64; ++c) {
    float x = row[c];
    sum += x; sq += x * x;
  }
  PART[(r * 8 + p) * 2 + 0] = sum;
  PART[(r * 8 + p) * 2 + 1] = sq;
  __syncthreads();
  if (p == 0) {
    float S = 0.0f, Q = 0.0f;
    for (int j = 0; j < 8; ++j) { S += PART[(r * 8 + j) * 2]; Q += PART[(r * 8 + j) * 2 + 1]; }
    float m = S * (1.0f / 512.0f);
    float var = Q * (1.0f / 512.0f) - m * m;
    STAT[r * 2 + 0] = m;
    STAT[r * 2 + 1] = rsqrtf(var + LN_EPS);
  }
  __syncthreads();
  for (int it = 0; it < 64; ++it) {
    int i = it * 256 + tid;
    int rr = i >> 9, c = i & 511;
    float x = (FB[i] - STAT[rr * 2]) * STAT[rr * 2 + 1] * s[c] + b[c];
    HB[i] = (bf16)(x > 0.0f ? x : 0.0f);
  }
  __syncthreads();
}

// stage a 32-K chunk of transposed weight wT[512][krow] into WC[512][KPAD]
static __device__ __forceinline__ void stage_wT(bf16* WC, const bf16* __restrict__ wT,
                                                int krow, int kc, int tid) {
#pragma unroll
  for (int it = 0; it < 8; ++it) {              // 512 cols x 4 k-octets = 2048 units
    int u = it * 256 + tid;
    int n = u >> 2, ko = (u & 3) << 3;
    *(v8bf*)(WC + n * KPAD + ko) = *(const v8bf*)(wT + (size_t)n * krow + kc + ko);
  }
}

__global__ void __launch_bounds__(256) encoder_kernel(
    const float* __restrict__ obs, const int* __restrict__ acts,
    const bf16* __restrict__ w0T, const float* __restrict__ b0,
    const float* __restrict__ ln0s, const float* __restrict__ ln0b,
    const bf16* __restrict__ w1T, const float* __restrict__ b1,
    const float* __restrict__ ln1s, const float* __restrict__ ln1b,
    const bf16* __restrict__ wirT, const float* __restrict__ bir,
    const bf16* __restrict__ wizT, const float* __restrict__ biz,
    const bf16* __restrict__ winT, const float* __restrict__ bin_,
    bf16* __restrict__ gir, bf16* __restrict__ giz, bf16* __restrict__ gin) {
  extern __shared__ char smem[];
  bf16*  A0   = (bf16*)(smem + ENC_A0_OFF);     // [32][32] row-major
  bf16*  WC   = (bf16*)(smem + ENC_WC_OFF);     // [512][KPAD] transposed
  float* FB   = (float*)(smem + ENC_F32_OFF);   // [32][512]
  bf16*  HB   = (bf16*)(smem + ENC_HB_OFF);     // [32][512] row-major
  float* PART = (float*)(smem + ENC_PART_OFF);
  float* STAT = (float*)(smem + ENC_STAT_OFF);

  const int tid = threadIdx.x;
  const int lane = tid & 31;
  const int wave = tid >> 5;
  const int wr = wave & 1;       // row half (16 rows)
  const int wcq = wave >> 1;     // col quarter (128 cols)
  const size_t R0 = (size_t)blockIdx.x * 32;
  const int hl = lane >> 4, ln = lane & 15;
  const int row0 = wr * 16 + 8 * hl;

  v8f acc[8];

  // ----- Layer 0: x[530] @ w0 -----
#pragma unroll
  for (int i = 0; i < 8; ++i) acc[i] = vzero8();
  for (int kc = 0; kc < OBSD; kc += 32) {
    if (tid < 128) {                            // 32 rows x 4 k-octets
      int r = tid >> 2, ko = (tid & 3) << 3;
      float f[8];
#pragma unroll
      for (int e = 0; e < 8; ++e) f[e] = obs[(R0 + r) * OBSD + kc + ko + e];
      *(v8bf*)(A0 + r * 32 + ko) = pack8(f);
    }
    stage_wT(WC, w0T, W0K, kc, tid);
    __syncthreads();
    v16bf a = load_a_frag(A0 + wr * 16 * 32, lane, 32);
    v16bf bfr[8];
#pragma unroll
    for (int ct = 0; ct < 8; ++ct)
      bfr[ct] = load_b_frag_tr(WC + (wcq * 128 + ct * 16) * KPAD, lane, KPAD);
#pragma unroll
    for (int ct = 0; ct < 8; ++ct) acc[ct] = wmma_bf16(a, bfr[ct], acc[ct]);
    __syncthreads();
  }
#pragma unroll
  for (int ct = 0; ct < 8; ++ct) {
    int col = wcq * 128 + ct * 16 + ln;
    float* fp = FB + row0 * 512 + col;
#pragma unroll
    for (int v = 0; v < 8; ++v) {
      int act = acts[R0 + row0 + v];
      fp[v * 512] = acc[ct][v] + b0[col] + (float)w0T[(size_t)col * W0K + OBSD + act];
    }
  }
  __syncthreads();
  ln_relu_block(FB, HB, PART, STAT, ln0s, ln0b, tid);

  // ----- Layer 1: h0 @ w1 -----
#pragma unroll
  for (int i = 0; i < 8; ++i) acc[i] = vzero8();
  for (int kc = 0; kc < HID; kc += 32) {
    stage_wT(WC, w1T, HID, kc, tid);
    __syncthreads();
    v16bf a = load_a_frag(HB + wr * 16 * 512 + kc, lane, 512);
    v16bf bfr[8];
#pragma unroll
    for (int ct = 0; ct < 8; ++ct)
      bfr[ct] = load_b_frag_tr(WC + (wcq * 128 + ct * 16) * KPAD, lane, KPAD);
#pragma unroll
    for (int ct = 0; ct < 8; ++ct) acc[ct] = wmma_bf16(a, bfr[ct], acc[ct]);
    __syncthreads();
  }
#pragma unroll
  for (int ct = 0; ct < 8; ++ct) {
    int col = wcq * 128 + ct * 16 + ln;
    float* fp = FB + row0 * 512 + col;
#pragma unroll
    for (int v = 0; v < 8; ++v) fp[v * 512] = acc[ct][v] + b1[col];
  }
  __syncthreads();
  ln_relu_block(FB, HB, PART, STAT, ln1s, ln1b, tid);

  // ----- Input gates: gi_g = h1 @ w_ig + b_ig -----
  for (int g = 0; g < 3; ++g) {
    const bf16* wg = (g == 0) ? wirT : (g == 1) ? wizT : winT;
    const float* bg = (g == 0) ? bir : (g == 1) ? biz : bin_;
    bf16* og = (g == 0) ? gir : (g == 1) ? giz : gin;
#pragma unroll
    for (int i = 0; i < 8; ++i) acc[i] = vzero8();
    for (int kc = 0; kc < HID; kc += 32) {
      stage_wT(WC, wg, HID, kc, tid);
      __syncthreads();
      v16bf a = load_a_frag(HB + wr * 16 * 512 + kc, lane, 512);
      v16bf bfr[8];
#pragma unroll
      for (int ct = 0; ct < 8; ++ct)
        bfr[ct] = load_b_frag_tr(WC + (wcq * 128 + ct * 16) * KPAD, lane, KPAD);
#pragma unroll
      for (int ct = 0; ct < 8; ++ct) acc[ct] = wmma_bf16(a, bfr[ct], acc[ct]);
      __syncthreads();
    }
#pragma unroll
    for (int ct = 0; ct < 8; ++ct) {
      int col = wcq * 128 + ct * 16 + ln;
      bf16* op = og + (R0 + row0) * HID + col;
#pragma unroll
      for (int v = 0; v < 8; ++v) op[v * HID] = (bf16)(acc[ct][v] + bg[col]);
    }
  }
}

// ---------------------------------------------------------------------------
// Persistent GRU scan: ONE kernel runs all T steps. 128 WGs (16 Btiles x
// 8 Htiles), each preloads its 3-gate weight column slice into LDS once
// (3 x 64 x WFK halves ~195KB, resident for all steps), then per step stages
// only the masked f32 hidden chunk. Steps are separated by a device-scope
// atomic grid barrier (monotonic counter).
// Dynamic LDS: WF = 3*64*WFK*2 = 199680 B ; AS = 2*32*32*2 = 4096 B
// ---------------------------------------------------------------------------
#define GRU_WF_BYTES (3 * 64 * WFK * 2)
#define GRU_SMEM_BYTES (GRU_WF_BYTES + 4096)

__global__ void __launch_bounds__(256) gru_scan_kernel(
    float* __restrict__ hA, float* __restrict__ hB,
    const unsigned char* __restrict__ dones_all,
    const bf16* __restrict__ whrT, const bf16* __restrict__ whzT, const bf16* __restrict__ whnT,
    const bf16* __restrict__ gir, const bf16* __restrict__ giz, const bf16* __restrict__ gin,
    const float* __restrict__ bhn,
    bf16* __restrict__ Y,
    unsigned* __restrict__ bar) {
  extern __shared__ char smem[];
  bf16* WF = (bf16*)smem;                       // [3][64][WFK] resident weights
  bf16* AS = (bf16*)(smem + GRU_WF_BYTES);      // [2][32][32] h chunk double buffer

  const int tid = threadIdx.x, lane = tid & 31, wave = tid >> 5;
  const int wr = wave >> 2;   // 0..1  (16-row half)
  const int wc2 = wave & 3;   // 0..3  (16-col slice)
  const int bt = blockIdx.x & 15, ht = blockIdx.x >> 4;
  const int rbase = bt * 32, cbase = ht * 64;
  const unsigned nwg = gridDim.x;

  // ---- preload this WG's weight column slice (3 gates x 64 cols x 512 k) ----
#pragma unroll
  for (int it = 0; it < 48; ++it) {             // 3*64*64 k-octets = 12288 units
    int u = it * 256 + tid;
    int g = u >> 12, j = u & 4095;
    int n = j >> 6, ko = (j & 63) << 3;
    const bf16* w = (g == 0) ? whrT : (g == 1) ? whzT : whnT;
    *(v8bf*)(WF + (g * 64 + n) * WFK + ko) = *(const v8bf*)(w + (size_t)(cbase + n) * HID + ko);
  }
  __syncthreads();

  const int hl = lane >> 4, ln = lane & 15;
  const int col = cbase + wc2 * 16 + ln;
  const int row0 = rbase + wr * 16 + 8 * hl;
  const float bh = bhn[col];

  for (int t = 0; t < T_STEPS; ++t) {
    const float* hp = (t & 1) ? hB : hA;
    float* hx = (t & 1) ? hA : hB;
    const unsigned char* dn = dones_all + (size_t)t * BATCH;

    v8f acc[3];
#pragma unroll
    for (int g = 0; g < 3; ++g) acc[g] = vzero8();

    auto stageA = [&](int buf, int kc) {
      if (tid < 128) {                          // 32 rows x 4 k-octets
        int r = tid >> 2, ko = (tid & 3) << 3;
        int row = rbase + r;
        bool d = dn[row] != 0;
        float f[8];
#pragma unroll
        for (int e = 0; e < 8; ++e)
          f[e] = d ? 0.0f : hp[(size_t)row * HID + kc + ko + e];
        *(v8bf*)(AS + (buf * 32 + r) * 32 + ko) = pack8(f);
      }
    };

    stageA(0, 0);
    __syncthreads();
    for (int i = 0; i < 16; ++i) {
      int cur = i & 1;
      if (i < 15) stageA(cur ^ 1, (i + 1) * 32);
      v16bf a = load_a_frag(AS + cur * 32 * 32 + wr * 16 * 32, lane, 32);
      v16bf bfr[3];
#pragma unroll
      for (int g = 0; g < 3; ++g)
        bfr[g] = load_b_frag_tr(WF + (g * 64 + wc2 * 16) * WFK + i * 32, lane, WFK);
#pragma unroll
      for (int g = 0; g < 3; ++g) acc[g] = wmma_bf16(a, bfr[g], acc[g]);
      __syncthreads();
    }

    // ---- epilogue: gates + blend (constant-offset addressing) ----
    {
      const size_t hoff = (size_t)row0 * HID + col;
      const size_t goff = ((size_t)t * BATCH + row0) * HID + col;
      const bf16* gr = gir + goff;
      const bf16* gz = giz + goff;
      const bf16* gnp = gin + goff;
      const float* hpp = hp + hoff;
      float* hxp = hx + hoff;
      bf16* yp = Y + goff;
      const unsigned long long dm = *(const unsigned long long*)(dn + row0);
#pragma unroll
      for (int v = 0; v < 8; ++v) {
        bool done = ((dm >> (8 * v)) & 0xffull) != 0;
        float r = sigmoidf_(acc[0][v] + (float)gr[v * HID]);
        float z = sigmoidf_(acc[1][v] + (float)gz[v * HID]);
        float n = tanhf((float)gnp[v * HID] + r * (acc[2][v] + bh));
        float hm = done ? 0.0f : hpp[v * HID];
        float hn = (1.0f - z) * n + z * hm;
        hxp[v * HID] = hn;
        yp[v * HID] = (bf16)hn;
      }
    }

    // ---- grid barrier (skip after last step) ----
    if (t < T_STEPS - 1) {
      __threadfence();          // each wave drains its own stores to device scope
      __syncthreads();
      if (tid == 0) {
        __hip_atomic_fetch_add(bar, 1u, __ATOMIC_ACQ_REL, __HIP_MEMORY_SCOPE_AGENT);
        const unsigned target = nwg * (unsigned)(t + 1);
        while (__hip_atomic_load(bar, __ATOMIC_ACQUIRE, __HIP_MEMORY_SCOPE_AGENT) < target)
          __builtin_amdgcn_s_sleep(1);
      }
      __syncthreads();
    }
  }
}

// ---------------------------------------------------------------------------
// Head: q = Y(bf16)[T*B,512] @ w_out[512,18] + b_out  (N padded to 32)
// ---------------------------------------------------------------------------
#define WSK 520
__global__ void __launch_bounds__(256) qvals_kernel(
    const bf16* __restrict__ Y, const float* __restrict__ wout,
    const float* __restrict__ bout, float* __restrict__ q) {
  __shared__ __attribute__((aligned(16))) bf16 Ws[32][WSK];
  const int tid = threadIdx.x, lane = tid & 31, wave = tid >> 5;
#pragma unroll
  for (int it = 0; it < 64; ++it) {
    int i = it * 256 + tid;
    int n = i & 31, k = i >> 5;
    Ws[n][k] = (n < ACTD) ? (bf16)wout[k * ACTD + n] : (bf16)0.0f;
  }
  __syncthreads();

  const size_t rowbase = (size_t)blockIdx.x * 128 + (size_t)wave * 16;
  v8f acc[2];
  acc[0] = vzero8(); acc[1] = vzero8();

  const int rA = lane & 15;
  const int koff = (lane >> 4) << 3;
  const bf16* arow = Y + (rowbase + rA) * HID;
  for (int kc = 0; kc < HID; kc += 32) {
    const bf16* p = arow + kc + koff;
    v16bf a = cat16(*(const v8bf*)(p), *(const v8bf*)(p + 16));
    v16bf b0f = load_b_frag_tr(&Ws[0][kc], lane, WSK);
    v16bf b1f = load_b_frag_tr(&Ws[16][kc], lane, WSK);
    acc[0] = wmma_bf16(a, b0f, acc[0]);
    acc[1] = wmma_bf16(a, b1f, acc[1]);
  }

  const int hl = lane >> 4, ln = lane & 15;
#pragma unroll
  for (int ct = 0; ct < 2; ++ct) {
    int col = ct * 16 + ln;
    if (col < ACTD) {
      float* qp = q + (rowbase + 8 * hl) * ACTD + col;
      float bo = bout[col];
#pragma unroll
      for (int v = 0; v < 8; ++v) qp[v * ACTD] = acc[ct][v] + bo;
    }
  }
}

// ---------------------------------------------------------------------------
// Host launcher
// ---------------------------------------------------------------------------
extern "C" void kernel_launch(void* const* d_in, const int* in_sizes, int n_in,
                              void* d_out, int out_size, void* d_ws, size_t ws_size,
                              hipStream_t stream) {
  (void)in_sizes; (void)n_in; (void)out_size; (void)ws_size;
  const float* hidden = (const float*)d_in[0];
  const float* obs    = (const float*)d_in[1];
  const unsigned char* dones = (const unsigned char*)d_in[2];
  const int*   acts   = (const int*)d_in[3];
  const float* w0  = (const float*)d_in[4];
  const float* b0  = (const float*)d_in[5];
  const float* ln0s = (const float*)d_in[6];
  const float* ln0b = (const float*)d_in[7];
  const float* w1  = (const float*)d_in[8];
  const float* b1  = (const float*)d_in[9];
  const float* ln1s = (const float*)d_in[10];
  const float* ln1b = (const float*)d_in[11];
  const float* wir = (const float*)d_in[12];
  const float* bir = (const float*)d_in[13];
  const float* wiz = (const float*)d_in[14];
  const float* biz = (const float*)d_in[15];
  const float* win = (const float*)d_in[16];
  const float* bin_ = (const float*)d_in[17];
  const float* whr = (const float*)d_in[18];
  const float* whz = (const float*)d_in[19];
  const float* whn = (const float*)d_in[20];
  const float* bhn = (const float*)d_in[21];
  const float* wout = (const float*)d_in[22];
  const float* bout = (const float*)d_in[23];

  char* ws = (char*)d_ws;
  const size_t S = (size_t)T_STEPS * BATCH * HID * sizeof(bf16);   // 64 MiB
  bf16* gir = (bf16*)(ws + 0 * S);
  bf16* giz = (bf16*)(ws + 1 * S);
  bf16* gin = (bf16*)(ws + 2 * S);
  bf16* Y   = (bf16*)(ws + 3 * S);
  float* hA = (float*)(ws + 4 * S);
  float* hB = (float*)(ws + 4 * S + (size_t)BATCH * HID * sizeof(float));
  char* wb  = ws + 4 * S + 2 * (size_t)BATCH * HID * sizeof(float);
  const size_t W0TB = (size_t)512 * W0K * sizeof(bf16);            // 557056
  const size_t WHB  = (size_t)512 * 512 * sizeof(bf16);            // 524288
  bf16* w0T  = (bf16*)(wb);
  bf16* w1T  = (bf16*)(wb + W0TB);
  bf16* wirT = (bf16*)(wb + W0TB + 1 * WHB);
  bf16* wizT = (bf16*)(wb + W0TB + 2 * WHB);
  bf16* winT = (bf16*)(wb + W0TB + 3 * WHB);
  bf16* whrT = (bf16*)(wb + W0TB + 4 * WHB);
  bf16* whzT = (bf16*)(wb + W0TB + 5 * WHB);
  bf16* whnT = (bf16*)(wb + W0TB + 6 * WHB);
  unsigned* bar = (unsigned*)(wb + W0TB + 7 * WHB);

  // 1) convert + transpose weights f32 -> bf16 (wT[n][k]); zero grid barrier
  {
    const int n0 = IN0 * 512, nh = 512 * 512;
    cvt_tr_kernel<<<(n0 + 255) / 256, 256, 0, stream>>>(w0, w0T, IN0, W0K);
    cvt_tr_kernel<<<(nh + 255) / 256, 256, 0, stream>>>(w1, w1T, HID, HID);
    cvt_tr_kernel<<<(nh + 255) / 256, 256, 0, stream>>>(wir, wirT, HID, HID);
    cvt_tr_kernel<<<(nh + 255) / 256, 256, 0, stream>>>(wiz, wizT, HID, HID);
    cvt_tr_kernel<<<(nh + 255) / 256, 256, 0, stream>>>(win, winT, HID, HID);
    cvt_tr_kernel<<<(nh + 255) / 256, 256, 0, stream>>>(whr, whrT, HID, HID);
    cvt_tr_kernel<<<(nh + 255) / 256, 256, 0, stream>>>(whz, whzT, HID, HID);
    cvt_tr_kernel<<<(nh + 255) / 256, 256, 0, stream>>>(whn, whnT, HID, HID);
    zero_u32_kernel<<<1, 64, 0, stream>>>(bar);
  }

  // 2) init hidden ping-pong buffer
  copy_f32_kernel<<<(BATCH * HID + 255) / 256, 256, 0, stream>>>(hidden, hA, BATCH * HID);

  // 3) fused encoder + input-gate GEMMs (batched over all T*B rows)
  encoder_kernel<<<(T_STEPS * BATCH) / 32, 256, ENC_SMEM_BYTES, stream>>>(
      obs, acts, w0T, b0, ln0s, ln0b, w1T, b1, ln1s, ln1b,
      wirT, bir, wizT, biz, winT, bin_, gir, giz, gin);

  // 4) persistent GRU scan (single launch, LDS-resident weights, grid barrier)
  gru_scan_kernel<<<GRU_NWG, 256, GRU_SMEM_BYTES, stream>>>(
      hA, hB, dones, whrT, whzT, whnT, gir, giz, gin, bhn, Y, bar);

  // 5) Q-value head over all T*B rows
  float* q_out = (float*)d_out + (size_t)BATCH * HID;
  qvals_kernel<<<(T_STEPS * BATCH) / 128, 256, 0, stream>>>(Y, wout, bout, q_out);

  // 6) new_hidden = h after last step (t=127 odd -> wrote hA)
  copy_f32_kernel<<<(BATCH * HID + 255) / 256, 256, 0, stream>>>(hA, (float*)d_out, BATCH * HID);
}
